// ROI_73486890434594
// MI455X (gfx1250) — matrix operand order repaired and checked by reading, not compiled, not run on hardware
//
#include <hip/hip_runtime.h>

// ---------------------------------------------------------------------------
// Types for CDNA5 WMMA (wave32, 16x16x32 bf16 -> f32 accumulate)
// ---------------------------------------------------------------------------
typedef __attribute__((ext_vector_type(16))) __bf16 v16bf;
typedef __attribute__((ext_vector_type(8)))  float  v8f;
typedef __attribute__((ext_vector_type(4)))  float  f32x4;

union Frag {
    v16bf bf;
    f32x4 f4[2];
};

__device__ __forceinline__ v8f wmma_bf16(v16bf a, v16bf b, v8f c) {
    return __builtin_amdgcn_wmma_f32_16x16x32_bf16(
        /*neg_a=*/false, a, /*neg_b=*/false, b,
        /*c_mod=*/(short)0, c, /*reuse_a=*/false, /*reuse_b=*/false);
}

__device__ __forceinline__ unsigned short f2bf(float x) {
    unsigned int u = __float_as_uint(x);
    unsigned int r = u + 0x7FFFu + ((u >> 16) & 1u);   // round-to-nearest-even
    return (unsigned short)(r >> 16);
}
__device__ __forceinline__ float bf2f(unsigned short h) {
    return __uint_as_float(((unsigned int)h) << 16);
}

#define NPROP 512
#define NCLS  81
#define NDET  100
#define HIDD  1024
#define K1    12544   // 256*7*7
#define FEATC 256
#define FEATH 200
#define FEATW 200

// ---------------------------------------------------------------------------
// fp32 -> bf16 conversion (weights)
// ---------------------------------------------------------------------------
__global__ void cvt_bf16_kernel(const float* __restrict__ in,
                                unsigned short* __restrict__ out, int n) {
    int t = blockIdx.x * blockDim.x + threadIdx.x;
    if (t < n) out[t] = f2bf(in[t]);
}

// ---------------------------------------------------------------------------
// ROI align (sr=2, scale=0.25). nhwc=0 -> (N, C*P*P) with (c,y,x) order (box
// head GEMM A matrix). nhwc=1 -> (N,P,P,C) channel-contiguous (mask head).
// ---------------------------------------------------------------------------
__global__ void roi_align_kernel(const float* __restrict__ feat,
                                 const float* __restrict__ rois,
                                 unsigned short* __restrict__ out,
                                 int Nrois, int P, int nhwc) {
    int t = blockIdx.x * blockDim.x + threadIdx.x;
    int total = Nrois * P * P;
    if (t >= total) return;
    int n = t / (P * P);
    int p = t - n * (P * P);
    int y = p / P;
    int x = p - y * P;

    float x1 = rois[n * 4 + 0] * 0.25f, y1 = rois[n * 4 + 1] * 0.25f;
    float x2 = rois[n * 4 + 2] * 0.25f, y2 = rois[n * 4 + 3] * 0.25f;
    float rw = fmaxf(x2 - x1, 1.0f), rh = fmaxf(y2 - y1, 1.0f);
    float bw = rw / (float)P, bh = rh / (float)P;

    int y0i[2], y1i[2], x0i[2], x1i[2];
    float ly[2], lx[2];
    for (int s = 0; s < 2; ++s) {
        float yy = y1 + ((float)y + ((float)s + 0.5f) * 0.5f) * bh;
        yy = fminf(fmaxf(yy, 0.0f), (float)(FEATH - 1));
        float yf = floorf(yy);
        y0i[s] = (int)yf;
        y1i[s] = (y0i[s] + 1 < FEATH - 1) ? y0i[s] + 1 : FEATH - 1;
        ly[s] = yy - yf;
        float xx = x1 + ((float)x + ((float)s + 0.5f) * 0.5f) * bw;
        xx = fminf(fmaxf(xx, 0.0f), (float)(FEATW - 1));
        float xf = floorf(xx);
        x0i[s] = (int)xf;
        x1i[s] = (x0i[s] + 1 < FEATW - 1) ? x0i[s] + 1 : FEATW - 1;
        lx[s] = xx - xf;
    }

    for (int c = 0; c < FEATC; ++c) {
        const float* fc = feat + (size_t)c * (FEATH * FEATW);
        float acc = 0.0f;
        for (int sy = 0; sy < 2; ++sy) {
            const float* r0 = fc + y0i[sy] * FEATW;
            const float* r1 = fc + y1i[sy] * FEATW;
            float wy = ly[sy];
            for (int sx = 0; sx < 2; ++sx) {
                float wx = lx[sx];
                float v0 = r0[x0i[sx]] * (1.0f - wx) + r0[x1i[sx]] * wx;
                float v1 = r1[x0i[sx]] * (1.0f - wx) + r1[x1i[sx]] * wx;
                acc += v0 * (1.0f - wy) + v1 * wy;
            }
        }
        acc *= 0.25f;
        size_t off = nhwc ? ((size_t)t * FEATC + c)
                          : ((size_t)n * (FEATC * P * P) + (size_t)c * (P * P) + y * P + x);
        out[off] = f2bf(acc);
    }
}

// ---------------------------------------------------------------------------
// Dense GEMM: C[M,N] = A[M,K](bf16) * B[N,K]^T(bf16) + bias, optional relu.
// Wave computes a 16x64 strip; block = 8 waves = 128 rows.
// Software-pipelined: body loads slice k+32 then consumes slice k; the final
// slice is consumed in an epilogue (no clamped index -> constant-stride
// addressing in the loop).
// ---------------------------------------------------------------------------
__global__ void gemm_bf16_kernel(const unsigned short* __restrict__ A,
                                 const unsigned short* __restrict__ B,
                                 const float* __restrict__ bias,
                                 float* __restrict__ outF,
                                 unsigned short* __restrict__ outBF,
                                 int M, int N, int K, int relu) {
    const int tid  = threadIdx.x;
    const int lane = tid & 31;
    const int wv   = tid >> 5;
    const int g    = lane >> 4;
    const int lr   = lane & 15;
    const int tile_m = blockIdx.x * 128 + wv * 16;
    const int tile_n = blockIdx.y * 64;

    int am = tile_m + lr; if (am >= M) am = M - 1;
    const unsigned short* Ap = A + (size_t)am * K;

    const unsigned short* Bp0, *Bp1, *Bp2, *Bp3;
    {
        int n0 = tile_n + 0 * 16 + lr; if (n0 >= N) n0 = N - 1;
        int n1 = tile_n + 1 * 16 + lr; if (n1 >= N) n1 = N - 1;
        int n2 = tile_n + 2 * 16 + lr; if (n2 >= N) n2 = N - 1;
        int n3 = tile_n + 3 * 16 + lr; if (n3 >= N) n3 = N - 1;
        Bp0 = B + (size_t)n0 * K; Bp1 = B + (size_t)n1 * K;
        Bp2 = B + (size_t)n2 * K; Bp3 = B + (size_t)n3 * K;
    }

    v8f acc0 = {0,0,0,0,0,0,0,0}, acc1 = {0,0,0,0,0,0,0,0};
    v8f acc2 = {0,0,0,0,0,0,0,0}, acc3 = {0,0,0,0,0,0,0,0};

    Frag a_c, b0_c, b1_c, b2_c, b3_c;
    {
        const int bo = g * 16;
        a_c.f4[0]  = *(const f32x4*)(Ap + g * 16);
        a_c.f4[1]  = *(const f32x4*)(Ap + g * 16 + 8);
        b0_c.f4[0] = *(const f32x4*)(Bp0 + bo); b0_c.f4[1] = *(const f32x4*)(Bp0 + bo + 8);
        b1_c.f4[0] = *(const f32x4*)(Bp1 + bo); b1_c.f4[1] = *(const f32x4*)(Bp1 + bo + 8);
        b2_c.f4[0] = *(const f32x4*)(Bp2 + bo); b2_c.f4[1] = *(const f32x4*)(Bp2 + bo + 8);
        b3_c.f4[0] = *(const f32x4*)(Bp3 + bo); b3_c.f4[1] = *(const f32x4*)(Bp3 + bo + 8);
    }

    for (int k0 = 0; k0 + 32 < K; k0 += 32) {
        const int kn = k0 + 32;
        // Speculative prefetch two slices ahead (no guard: prefetch faults are
        // dropped silently; overshoot stays inside the workspace).
        __builtin_prefetch(Ap + kn + 32, 0, 1);
        __builtin_prefetch(Bp0 + kn + 32, 0, 1);

        Frag a_n, b0_n, b1_n, b2_n, b3_n;
        a_n.f4[0]  = *(const f32x4*)(Ap + kn + g * 16);
        a_n.f4[1]  = *(const f32x4*)(Ap + kn + g * 16 + 8);
        const int bo = kn + g * 16;
        b0_n.f4[0] = *(const f32x4*)(Bp0 + bo); b0_n.f4[1] = *(const f32x4*)(Bp0 + bo + 8);
        b1_n.f4[0] = *(const f32x4*)(Bp1 + bo); b1_n.f4[1] = *(const f32x4*)(Bp1 + bo + 8);
        b2_n.f4[0] = *(const f32x4*)(Bp2 + bo); b2_n.f4[1] = *(const f32x4*)(Bp2 + bo + 8);
        b3_n.f4[0] = *(const f32x4*)(Bp3 + bo); b3_n.f4[1] = *(const f32x4*)(Bp3 + bo + 8);

        acc0 = wmma_bf16(a_c.bf, b0_c.bf, acc0);
        acc1 = wmma_bf16(a_c.bf, b1_c.bf, acc1);
        acc2 = wmma_bf16(a_c.bf, b2_c.bf, acc2);
        acc3 = wmma_bf16(a_c.bf, b3_c.bf, acc3);

        a_c = a_n; b0_c = b0_n; b1_c = b1_n; b2_c = b2_n; b3_c = b3_n;
    }
    // Epilogue: consume the last slice.
    acc0 = wmma_bf16(a_c.bf, b0_c.bf, acc0);
    acc1 = wmma_bf16(a_c.bf, b1_c.bf, acc1);
    acc2 = wmma_bf16(a_c.bf, b2_c.bf, acc2);
    acc3 = wmma_bf16(a_c.bf, b3_c.bf, acc3);

    v8f accs[4] = {acc0, acc1, acc2, acc3};
    for (int j = 0; j < 4; ++j) {
        int n = tile_n + j * 16 + lr;
        if (n >= N) continue;
        float bv = bias ? bias[n] : 0.0f;
#pragma unroll
        for (int r = 0; r < 8; ++r) {
            int m = tile_m + r + g * 8;
            if (m >= M) continue;
            float v = accs[j][r] + bv;
            if (relu) v = fmaxf(v, 0.0f);
            size_t off = (size_t)m * N + n;
            if (outF)  outF[off]  = v;
            if (outBF) outBF[off] = f2bf(v);
        }
    }
}

// ---------------------------------------------------------------------------
// Implicit-GEMM 3x3 conv, pad=1, NHWC(14,14,256) -> NHWC(14,14,256), relu.
// K ordered as (ky,kx,c): runs of 8 channels stay contiguous in memory.
// ---------------------------------------------------------------------------
__device__ __forceinline__ f32x4 conv_loadA(const unsigned short* __restrict__ In,
                                            int roi, int y, int x, int kk) {
    int tap = kk >> 8;
    int c0  = kk & 255;
    int ky = tap / 3, kx = tap - ky * 3;
    int iy = y + ky - 1, ix = x + kx - 1;
    if ((unsigned)iy >= 14u || (unsigned)ix >= 14u) {
        f32x4 z = {0.f, 0.f, 0.f, 0.f};
        return z;
    }
    return *(const f32x4*)(In + (((size_t)((roi * 14 + iy) * 14 + ix)) << 8) + c0);
}

__global__ void conv3x3_gemm_kernel(const unsigned short* __restrict__ In,
                                    const unsigned short* __restrict__ Wr, // (256,2304)
                                    const float* __restrict__ bias,
                                    unsigned short* __restrict__ Out,
                                    int M) {  // M = rois*196
    const int K = 2304, N = 256;
    const int tid  = threadIdx.x;
    const int lane = tid & 31;
    const int wv   = tid >> 5;
    const int g    = lane >> 4;
    const int lr   = lane & 15;
    const int tile_m = blockIdx.x * 128 + wv * 16;
    const int tile_n = blockIdx.y * 64;

    int am = tile_m + lr; if (am >= M) am = M - 1;
    int roi = am / 196;
    int p   = am - roi * 196;
    int y   = p / 14;
    int x   = p - y * 14;

    const unsigned short* Bp0 = Wr + (size_t)(tile_n + 0 * 16 + lr) * K;
    const unsigned short* Bp1 = Wr + (size_t)(tile_n + 1 * 16 + lr) * K;
    const unsigned short* Bp2 = Wr + (size_t)(tile_n + 2 * 16 + lr) * K;
    const unsigned short* Bp3 = Wr + (size_t)(tile_n + 3 * 16 + lr) * K;

    v8f acc0 = {0,0,0,0,0,0,0,0}, acc1 = {0,0,0,0,0,0,0,0};
    v8f acc2 = {0,0,0,0,0,0,0,0}, acc3 = {0,0,0,0,0,0,0,0};

    Frag a_c, b0_c, b1_c, b2_c, b3_c;
    {
        const int bo = g * 16;
        a_c.f4[0]  = conv_loadA(In, roi, y, x, g * 16);
        a_c.f4[1]  = conv_loadA(In, roi, y, x, g * 16 + 8);
        b0_c.f4[0] = *(const f32x4*)(Bp0 + bo); b0_c.f4[1] = *(const f32x4*)(Bp0 + bo + 8);
        b1_c.f4[0] = *(const f32x4*)(Bp1 + bo); b1_c.f4[1] = *(const f32x4*)(Bp1 + bo + 8);
        b2_c.f4[0] = *(const f32x4*)(Bp2 + bo); b2_c.f4[1] = *(const f32x4*)(Bp2 + bo + 8);
        b3_c.f4[0] = *(const f32x4*)(Bp3 + bo); b3_c.f4[1] = *(const f32x4*)(Bp3 + bo + 8);
    }

    for (int k0 = 0; k0 + 32 < K; k0 += 32) {
        const int kn = k0 + 32;
        Frag a_n, b0_n, b1_n, b2_n, b3_n;
        a_n.f4[0]  = conv_loadA(In, roi, y, x, kn + g * 16);
        a_n.f4[1]  = conv_loadA(In, roi, y, x, kn + g * 16 + 8);
        const int bo = kn + g * 16;
        b0_n.f4[0] = *(const f32x4*)(Bp0 + bo); b0_n.f4[1] = *(const f32x4*)(Bp0 + bo + 8);
        b1_n.f4[0] = *(const f32x4*)(Bp1 + bo); b1_n.f4[1] = *(const f32x4*)(Bp1 + bo + 8);
        b2_n.f4[0] = *(const f32x4*)(Bp2 + bo); b2_n.f4[1] = *(const f32x4*)(Bp2 + bo + 8);
        b3_n.f4[0] = *(const f32x4*)(Bp3 + bo); b3_n.f4[1] = *(const f32x4*)(Bp3 + bo + 8);

        acc0 = wmma_bf16(a_c.bf, b0_c.bf, acc0);
        acc1 = wmma_bf16(a_c.bf, b1_c.bf, acc1);
        acc2 = wmma_bf16(a_c.bf, b2_c.bf, acc2);
        acc3 = wmma_bf16(a_c.bf, b3_c.bf, acc3);

        a_c = a_n; b0_c = b0_n; b1_c = b1_n; b2_c = b2_n; b3_c = b3_n;
    }
    acc0 = wmma_bf16(a_c.bf, b0_c.bf, acc0);
    acc1 = wmma_bf16(a_c.bf, b1_c.bf, acc1);
    acc2 = wmma_bf16(a_c.bf, b2_c.bf, acc2);
    acc3 = wmma_bf16(a_c.bf, b3_c.bf, acc3);

    v8f accs[4] = {acc0, acc1, acc2, acc3};
    for (int j = 0; j < 4; ++j) {
        int n = tile_n + j * 16 + lr;
        float bv = bias[n];
#pragma unroll
        for (int r = 0; r < 8; ++r) {
            int m = tile_m + r + g * 8;
            if (m >= M) continue;
            float v = fmaxf(accs[j][r] + bv, 0.0f);
            Out[(size_t)m * N + n] = f2bf(v);
        }
    }
}

// ---------------------------------------------------------------------------
// Deconv 2x2 stride 2 as GEMM: A[19600,256] * Wd[1024,256]^T, where
// n = (dy*2+dx)*256 + o; scatter-store relu(out) to NHWC (100,28,28,256).
// ---------------------------------------------------------------------------
__global__ void deconv_gemm_kernel(const unsigned short* __restrict__ In,
                                   const unsigned short* __restrict__ Wd,
                                   const float* __restrict__ bias,  // [256] per o
                                   unsigned short* __restrict__ Out,
                                   int M) {
    const int K = 256;
    const int tid  = threadIdx.x;
    const int lane = tid & 31;
    const int wv   = tid >> 5;
    const int g    = lane >> 4;
    const int lr   = lane & 15;
    const int tile_m = blockIdx.x * 128 + wv * 16;
    const int tile_n = blockIdx.y * 64;

    int am = tile_m + lr; if (am >= M) am = M - 1;
    const unsigned short* Ap  = In + (size_t)am * K;
    const unsigned short* Bp0 = Wd + (size_t)(tile_n + 0 * 16 + lr) * K;
    const unsigned short* Bp1 = Wd + (size_t)(tile_n + 1 * 16 + lr) * K;
    const unsigned short* Bp2 = Wd + (size_t)(tile_n + 2 * 16 + lr) * K;
    const unsigned short* Bp3 = Wd + (size_t)(tile_n + 3 * 16 + lr) * K;

    v8f acc0 = {0,0,0,0,0,0,0,0}, acc1 = {0,0,0,0,0,0,0,0};
    v8f acc2 = {0,0,0,0,0,0,0,0}, acc3 = {0,0,0,0,0,0,0,0};

    Frag a_c, b0_c, b1_c, b2_c, b3_c;
    {
        const int bo = g * 16;
        a_c.f4[0]  = *(const f32x4*)(Ap + g * 16);
        a_c.f4[1]  = *(const f32x4*)(Ap + g * 16 + 8);
        b0_c.f4[0] = *(const f32x4*)(Bp0 + bo); b0_c.f4[1] = *(const f32x4*)(Bp0 + bo + 8);
        b1_c.f4[0] = *(const f32x4*)(Bp1 + bo); b1_c.f4[1] = *(const f32x4*)(Bp1 + bo + 8);
        b2_c.f4[0] = *(const f32x4*)(Bp2 + bo); b2_c.f4[1] = *(const f32x4*)(Bp2 + bo + 8);
        b3_c.f4[0] = *(const f32x4*)(Bp3 + bo); b3_c.f4[1] = *(const f32x4*)(Bp3 + bo + 8);
    }

    for (int k0 = 0; k0 + 32 < K; k0 += 32) {
        const int kn = k0 + 32;
        Frag a_n, b0_n, b1_n, b2_n, b3_n;
        a_n.f4[0]  = *(const f32x4*)(Ap + kn + g * 16);
        a_n.f4[1]  = *(const f32x4*)(Ap + kn + g * 16 + 8);
        const int bo = kn + g * 16;
        b0_n.f4[0] = *(const f32x4*)(Bp0 + bo); b0_n.f4[1] = *(const f32x4*)(Bp0 + bo + 8);
        b1_n.f4[0] = *(const f32x4*)(Bp1 + bo); b1_n.f4[1] = *(const f32x4*)(Bp1 + bo + 8);
        b2_n.f4[0] = *(const f32x4*)(Bp2 + bo); b2_n.f4[1] = *(const f32x4*)(Bp2 + bo + 8);
        b3_n.f4[0] = *(const f32x4*)(Bp3 + bo); b3_n.f4[1] = *(const f32x4*)(Bp3 + bo + 8);

        acc0 = wmma_bf16(a_c.bf, b0_c.bf, acc0);
        acc1 = wmma_bf16(a_c.bf, b1_c.bf, acc1);
        acc2 = wmma_bf16(a_c.bf, b2_c.bf, acc2);
        acc3 = wmma_bf16(a_c.bf, b3_c.bf, acc3);

        a_c = a_n; b0_c = b0_n; b1_c = b1_n; b2_c = b2_n; b3_c = b3_n;
    }
    acc0 = wmma_bf16(a_c.bf, b0_c.bf, acc0);
    acc1 = wmma_bf16(a_c.bf, b1_c.bf, acc1);
    acc2 = wmma_bf16(a_c.bf, b2_c.bf, acc2);
    acc3 = wmma_bf16(a_c.bf, b3_c.bf, acc3);

    v8f accs[4] = {acc0, acc1, acc2, acc3};
    for (int j = 0; j < 4; ++j) {
        int n  = tile_n + j * 16 + lr;   // < 1024 by construction
        int tp = n >> 8;
        int o  = n & 255;
        int dy = tp >> 1, dx = tp & 1;
        float bv = bias[o];
#pragma unroll
        for (int r = 0; r < 8; ++r) {
            int m = tile_m + r + g * 8;
            if (m >= M) continue;
            int roi = m / 196;
            int pp  = m - roi * 196;
            int yy  = pp / 14;
            int xx  = pp - yy * 14;
            float v = fmaxf(accs[j][r] + bv, 0.0f);
            Out[(((size_t)(roi * 28 + 2 * yy + dy)) * 28 + (2 * xx + dx)) * 256 + o] = f2bf(v);
        }
    }
}

// ---------------------------------------------------------------------------
// Weight reorders (fp32 -> bf16, GEMM layouts)
// ---------------------------------------------------------------------------
__global__ void reorder_conv_w_kernel(const float* __restrict__ w,  // OIHW (256,256,3,3)
                                      unsigned short* __restrict__ out) { // (256,2304)
    int t = blockIdx.x * blockDim.x + threadIdx.x;
    if (t >= 256 * 2304) return;
    int o = t / 2304;
    int k = t - o * 2304;
    int tap = k >> 8, c = k & 255;
    out[t] = f2bf(w[(size_t)(o * 256 + c) * 9 + tap]);
}

__global__ void reorder_deconv_w_kernel(const float* __restrict__ w,  // IOHW (256,256,2,2)
                                        unsigned short* __restrict__ out) { // (1024,256)
    int t = blockIdx.x * blockDim.x + threadIdx.x;
    if (t >= 1024 * 256) return;
    int n = t >> 8, i = t & 255;
    int tap = n >> 8, o = n & 255;
    out[t] = f2bf(w[(size_t)(i * 256 + o) * 4 + tap]);
}

// ---------------------------------------------------------------------------
// Softmax over 81 classes
// ---------------------------------------------------------------------------
__global__ void softmax_kernel(const float* __restrict__ logits,
                               float* __restrict__ scores) {
    int n = blockIdx.x * blockDim.x + threadIdx.x;
    if (n >= NPROP) return;
    const float* r = logits + (size_t)n * NCLS;
    float* o = scores + (size_t)n * NCLS;
    float mx = r[0];
    for (int c = 1; c < NCLS; ++c) mx = fmaxf(mx, r[c]);
    float s = 0.0f;
    for (int c = 0; c < NCLS; ++c) { float e = expf(r[c] - mx); o[c] = e; s += e; }
    float inv = 1.0f / s;
    for (int c = 0; c < NCLS; ++c) o[c] *= inv;
}

// ---------------------------------------------------------------------------
// Box decode (all 81 classes)
// ---------------------------------------------------------------------------
__global__ void decode_kernel(const float* __restrict__ bbox,
                              const float* __restrict__ reg,
                              float* __restrict__ boxes_dec) {
    int t = blockIdx.x * blockDim.x + threadIdx.x;
    if (t >= NPROP * NCLS) return;
    int n = t / NCLS, c = t - n * NCLS;
    float p0 = bbox[n * 4 + 0], p1 = bbox[n * 4 + 1];
    float p2 = bbox[n * 4 + 2], p3 = bbox[n * 4 + 3];
    float w = p2 - p0 + 1.0f, h = p3 - p1 + 1.0f;
    float cx = p0 + 0.5f * w, cy = p1 + 0.5f * h;
    const float* rr = reg + (size_t)n * (NCLS * 4) + c * 4;
    const float CL = 4.135166556742356f;  // log(1000/16)
    float dx = rr[0] * 0.1f;
    float dy = rr[1] * 0.1f;
    float dw = fminf(rr[2] * 0.2f, CL);
    float dh = fminf(rr[3] * 0.2f, CL);
    float pcx = dx * w + cx, pcy = dy * h + cy;
    float pw = expf(dw) * w, ph = expf(dh) * h;
    float* o = boxes_dec + (size_t)t * 4;
    o[0] = fminf(fmaxf(pcx - 0.5f * pw, 0.0f), 799.0f);
    o[1] = fminf(fmaxf(pcy - 0.5f * ph, 0.0f), 799.0f);
    o[2] = fminf(fmaxf(pcx + 0.5f * pw - 1.0f, 0.0f), 799.0f);
    o[3] = fminf(fmaxf(pcy + 0.5f * ph - 1.0f, 0.0f), 799.0f);
}

// ---------------------------------------------------------------------------
// Per-class greedy NMS: bitonic sort (512) in LDS, sequential suppress,
// write masked scores (80,512).
// ---------------------------------------------------------------------------
__device__ __forceinline__ bool det_before(float as, int ai, float bs, int bi) {
    return (as > bs) || (as == bs && ai < bi);
}

__global__ void nms_kernel(const float* __restrict__ scores,    // (512,81)
                           const float* __restrict__ boxes_dec, // (512,81,4)
                           float* __restrict__ masked) {        // (80,512)
    const int cls = blockIdx.x + 1;  // skip background
    const int tid = threadIdx.x;
    __shared__ float ss[NPROP];
    __shared__ int   si[NPROP];
    __shared__ float bx1[NPROP], by1[NPROP], bx2[NPROP], by2[NPROP], bar[NPROP];
    __shared__ int   keep[NPROP];
    __shared__ int   sup;

    for (int i = tid; i < NPROP; i += 256) {
        ss[i] = scores[(size_t)i * NCLS + cls];
        si[i] = i;
    }
    __syncthreads();

    for (int k = 2; k <= NPROP; k <<= 1) {
        for (int j = k >> 1; j > 0; j >>= 1) {
            for (int i = tid; i < NPROP; i += 256) {
                int ixj = i ^ j;
                if (ixj > i) {
                    bool up = ((i & k) == 0);
                    float s1 = ss[i], s2 = ss[ixj];
                    int   i1 = si[i], i2 = si[ixj];
                    bool sw = up ? det_before(s2, i2, s1, i1)
                                 : det_before(s1, i1, s2, i2);
                    if (sw) { ss[i] = s2; si[i] = i2; ss[ixj] = s1; si[ixj] = i1; }
                }
            }
            __syncthreads();
        }
    }

    for (int i = tid; i < NPROP; i += 256) {
        const float* b = boxes_dec + ((size_t)si[i] * NCLS + cls) * 4;
        bx1[i] = b[0]; by1[i] = b[1]; bx2[i] = b[2]; by2[i] = b[3];
        bar[i] = (b[2] - b[0] + 1.0f) * (b[3] - b[1] + 1.0f);
        keep[i] = 0;
    }
    __syncthreads();

    for (int i = 0; i < NPROP; ++i) {
        if (tid == 0) sup = 0;
        __syncthreads();
        for (int j = tid; j < i; j += 256) {
            if (keep[j]) {
                float xx1 = fmaxf(bx1[i], bx1[j]);
                float yy1 = fmaxf(by1[i], by1[j]);
                float xx2 = fminf(bx2[i], bx2[j]);
                float yy2 = fminf(by2[i], by2[j]);
                float w = fmaxf(0.0f, xx2 - xx1 + 1.0f);
                float h = fmaxf(0.0f, yy2 - yy1 + 1.0f);
                float inter = w * h;
                float iou = inter / (bar[i] + bar[j] - inter);
                if (iou > 0.5f) sup = 1;
            }
        }
        __syncthreads();
        if (tid == 0) keep[i] = !sup;
        __syncthreads();
    }

    for (int i = tid; i < NPROP; i += 256) {
        float s = ss[i];
        masked[(size_t)blockIdx.x * NPROP + si[i]] =
            (keep[i] && s > 0.05f) ? s : -1.0f;
    }
}

// ---------------------------------------------------------------------------
// Top-100 (single block), writes det boxes/labels/scores + ws copies.
// ---------------------------------------------------------------------------
__global__ void topk_kernel(float* __restrict__ masked,          // (80*512), mutated
                            const float* __restrict__ boxes_dec, // (512,81,4)
                            float* __restrict__ out,             // d_out
                            float* __restrict__ det_boxes_ws,
                            int* __restrict__ labels_ws) {
    const int tid = threadIdx.x;
    const int TOT = 80 * NPROP;
    __shared__ float bv[256];
    __shared__ int   bi[256];
    __shared__ float kv[NDET];
    __shared__ int   ki[NDET];

    for (int k = 0; k < NDET; ++k) {
        float best = -1e30f; int besti = TOT;
        for (int i = tid; i < TOT; i += 256) {
            float v = masked[i];
            if (v > best || (v == best && i < besti)) { best = v; besti = i; }
        }
        bv[tid] = best; bi[tid] = besti;
        __syncthreads();
        for (int s = 128; s > 0; s >>= 1) {
            if (tid < s) {
                if (bv[tid + s] > bv[tid] ||
                    (bv[tid + s] == bv[tid] && bi[tid + s] < bi[tid])) {
                    bv[tid] = bv[tid + s]; bi[tid] = bi[tid + s];
                }
            }
            __syncthreads();
        }
        if (tid == 0) {
            kv[k] = bv[0]; ki[k] = bi[0];
            masked[bi[0]] = -1e30f;
        }
        __syncthreads();
    }

    if (tid < NDET) {
        int idx = ki[tid];
        int c = idx / NPROP, n = idx - c * NPROP;
        int label = c + 1;
        const float* b = boxes_dec + ((size_t)n * NCLS + label) * 4;
        for (int j = 0; j < 4; ++j) {
            out[tid * 4 + j] = b[j];
            det_boxes_ws[tid * 4 + j] = b[j];
        }
        out[400 + tid]   = (float)label;
        out[78900 + tid] = fmaxf(kv[tid], 0.0f);
        labels_ws[tid]   = label;
    }
}

// ---------------------------------------------------------------------------
// Final mask: per (det, pixel) dot with mpred_w[label], sigmoid.
// ---------------------------------------------------------------------------
__global__ void mask_kernel(const unsigned short* __restrict__ dec_out, // (100,28,28,256)
                            const int* __restrict__ labels_ws,
                            const float* __restrict__ mpred_w,  // (81,256)
                            const float* __restrict__ mpred_b,  // (81)
                            float* __restrict__ out) {
    int t = blockIdx.x * blockDim.x + threadIdx.x;
    if (t >= NDET * 784) return;
    int det = t / 784;
    int label = labels_ws[det];
    const unsigned short* in = dec_out + (size_t)t * 256;
    const float* wr = mpred_w + (size_t)label * 256;
    float s = mpred_b[label];
    for (int c = 0; c < 256; ++c) s += bf2f(in[c]) * wr[c];
    out[500 + t] = 1.0f / (1.0f + expf(-s));
}

// ---------------------------------------------------------------------------
// Host launcher
// ---------------------------------------------------------------------------
extern "C" void kernel_launch(void* const* d_in, const int* in_sizes, int n_in,
                              void* d_out, int out_size, void* d_ws, size_t ws_size,
                              hipStream_t stream) {
    (void)in_sizes; (void)n_in; (void)out_size; (void)ws_size;
    const float* features = (const float*)d_in[0];
    const float* bbox     = (const float*)d_in[1];
    const float* fc1_w    = (const float*)d_in[3];
    const float* fc1_b    = (const float*)d_in[4];
    const float* fc2_w    = (const float*)d_in[5];
    const float* fc2_b    = (const float*)d_in[6];
    const float* cls_w    = (const float*)d_in[7];
    const float* cls_b    = (const float*)d_in[8];
    const float* reg_w    = (const float*)d_in[9];
    const float* reg_b    = (const float*)d_in[10];
    const float* mcw[4]   = {(const float*)d_in[11], (const float*)d_in[13],
                             (const float*)d_in[15], (const float*)d_in[17]};
    const float* mcb[4]   = {(const float*)d_in[12], (const float*)d_in[14],
                             (const float*)d_in[16], (const float*)d_in[18]};
    const float* deconv_w = (const float*)d_in[19];
    const float* deconv_b = (const float*)d_in[20];
    const float* mpred_w  = (const float*)d_in[21];
    const float* mpred_b  = (const float*)d_in[22];
    float* out = (float*)d_out;

    char* cur = (char*)d_ws;
    auto alloc = [&](size_t bytes) -> void* {
        void* p = (void*)cur;
        cur += (bytes + 255) & ~(size_t)255;
        return p;
    };
    unsigned short* pooled    = (unsigned short*)alloc((size_t)NPROP * K1 * 2);
    unsigned short* fc1_wb    = (unsigned short*)alloc((size_t)HIDD * K1 * 2);
    unsigned short* fc2_wb    = (unsigned short*)alloc((size_t)HIDD * HIDD * 2);
    unsigned short* cls_wb    = (unsigned short*)alloc((size_t)NCLS * HIDD * 2);
    unsigned short* reg_wb    = (unsigned short*)alloc((size_t)NCLS * 4 * HIDD * 2);
    unsigned short* xf1       = (unsigned short*)alloc((size_t)NPROP * HIDD * 2);
    unsigned short* xf2       = (unsigned short*)alloc((size_t)NPROP * HIDD * 2);
    float* cls_logits         = (float*)alloc((size_t)NPROP * NCLS * 4);
    float* box_reg            = (float*)alloc((size_t)NPROP * NCLS * 4 * 4);
    float* scores             = (float*)alloc((size_t)NPROP * NCLS * 4);
    float* boxes_dec          = (float*)alloc((size_t)NPROP * NCLS * 4 * 4);
    float* masked             = (float*)alloc((size_t)80 * NPROP * 4);
    float* det_boxes_ws       = (float*)alloc((size_t)NDET * 4 * 4);
    int*   labels_ws          = (int*)alloc((size_t)NDET * 4);
    unsigned short* mpool     = (unsigned short*)alloc((size_t)NDET * 196 * 256 * 2);
    unsigned short* mbufA     = (unsigned short*)alloc((size_t)NDET * 196 * 256 * 2);
    unsigned short* mbufB     = (unsigned short*)alloc((size_t)NDET * 196 * 256 * 2);
    unsigned short* convWr    = (unsigned short*)alloc((size_t)256 * 2304 * 2);
    unsigned short* deconvWr  = (unsigned short*)alloc((size_t)1024 * 256 * 2);
    unsigned short* dec_out   = (unsigned short*)alloc((size_t)NDET * 784 * 256 * 2);

    const int B = 256;
    // Weight conversions (bf16)
    cvt_bf16_kernel<<<(HIDD * K1 + B - 1) / B, B, 0, stream>>>(fc1_w, fc1_wb, HIDD * K1);
    cvt_bf16_kernel<<<(HIDD * HIDD + B - 1) / B, B, 0, stream>>>(fc2_w, fc2_wb, HIDD * HIDD);
    cvt_bf16_kernel<<<(NCLS * HIDD + B - 1) / B, B, 0, stream>>>(cls_w, cls_wb, NCLS * HIDD);
    cvt_bf16_kernel<<<(NCLS * 4 * HIDD + B - 1) / B, B, 0, stream>>>(reg_w, reg_wb, NCLS * 4 * HIDD);

    // Box head
    roi_align_kernel<<<(NPROP * 49 + B - 1) / B, B, 0, stream>>>(features, bbox, pooled, NPROP, 7, 0);
    gemm_bf16_kernel<<<dim3(4, 16), B, 0, stream>>>(pooled, fc1_wb, fc1_b, nullptr, xf1, NPROP, HIDD, K1, 1);
    gemm_bf16_kernel<<<dim3(4, 16), B, 0, stream>>>(xf1, fc2_wb, fc2_b, nullptr, xf2, NPROP, HIDD, HIDD, 1);
    gemm_bf16_kernel<<<dim3(4, 2),  B, 0, stream>>>(xf2, cls_wb, cls_b, cls_logits, nullptr, NPROP, NCLS, HIDD, 0);
    gemm_bf16_kernel<<<dim3(4, 6),  B, 0, stream>>>(xf2, reg_wb, reg_b, box_reg, nullptr, NPROP, NCLS * 4, HIDD, 0);

    // Postprocess
    softmax_kernel<<<2, B, 0, stream>>>(cls_logits, scores);
    decode_kernel<<<(NPROP * NCLS + B - 1) / B, B, 0, stream>>>(bbox, box_reg, boxes_dec);
    nms_kernel<<<80, B, 0, stream>>>(scores, boxes_dec, masked);
    topk_kernel<<<1, B, 0, stream>>>(masked, boxes_dec, out, det_boxes_ws, labels_ws);

    // Mask head
    const int Mm = NDET * 196;  // 19600
    roi_align_kernel<<<(Mm + B - 1) / B, B, 0, stream>>>(features, det_boxes_ws, mpool, NDET, 14, 1);

    unsigned short* cin[4]  = {mpool, mbufA, mbufB, mbufA};
    unsigned short* cout[4] = {mbufA, mbufB, mbufA, mbufB};
    for (int l = 0; l < 4; ++l) {
        reorder_conv_w_kernel<<<(256 * 2304 + B - 1) / B, B, 0, stream>>>(mcw[l], convWr);
        conv3x3_gemm_kernel<<<dim3((Mm + 127) / 128, 4), B, 0, stream>>>(cin[l], convWr, mcb[l], cout[l], Mm);
    }
    reorder_deconv_w_kernel<<<(1024 * 256 + B - 1) / B, B, 0, stream>>>(deconv_w, deconvWr);
    deconv_gemm_kernel<<<dim3((Mm + 127) / 128, 16), B, 0, stream>>>(mbufB, deconvWr, deconv_b, dec_out, Mm);

    mask_kernel<<<(NDET * 784 + B - 1) / B, B, 0, stream>>>(dec_out, labels_ws, mpred_w, mpred_b, out);
}
